// DeepseekV3MoE_52673478918593
// MI455X (gfx1250) — matrix operand order repaired and checked by reading, not compiled, not run on hardware
//
#include <hip/hip_runtime.h>
#include <hip/hip_bf16.h>
#include <math.h>

// ---------------------------------------------------------------------------
// DeepSeek-V3 MoE (dense-expert variant) for MI455X / gfx1250, wave32 + WMMA.
// H=1024, M=512, E=16, N=2048 tokens, shared expert 2M=1024.
// Register-blocked bf16 WMMA GEMMs; combine weights folded into inter so the
// down-projection is one long-K accumulation GEMM (K = 16*512 + 1024 = 9216).
// ---------------------------------------------------------------------------

typedef __bf16 bf16_t;
typedef __attribute__((ext_vector_type(16))) __bf16 v16bf;
typedef __attribute__((ext_vector_type(8)))  __bf16 v8bf;
typedef __attribute__((ext_vector_type(8)))  float  v8f;

#define HDIM   1024
#define MDIM   512
#define NEXP   16
#define NTOK   2048
#define TWOM   1024
#define SCALEF 2.5f

// ------------------------------- workspace layout (bytes) ------------------
#define OFF_X       ((size_t)0)                         // 2048*1024 bf16 = 4 MiB
#define OFF_WG      ((size_t)4194304)                   // 16*1024*512 bf16 = 16 MiB
#define OFF_WU      ((size_t)20971520)
#define OFF_WD      ((size_t)37748736)
#define OFF_SWG     ((size_t)54525952)                  // 1024*1024 bf16 = 2 MiB
#define OFF_SWU     ((size_t)56623104)
#define OFF_SWD     ((size_t)58720256)
#define OFF_INTER   ((size_t)60817408)                  // 16*2048*512 bf16 = 32 MiB
#define OFF_INTERS  ((size_t)94371840)                  // 2048*1024 bf16 = 4 MiB
#define OFF_COMB    ((size_t)98566144)                  // 2048*16 f32 = 128 KiB

// ---------------------------------------------------------------------------
// f32 -> bf16 elementwise conversion (grid-stride)
// ---------------------------------------------------------------------------
__global__ void k_convert(const float* __restrict__ in, bf16_t* __restrict__ out, int n) {
    int i = blockIdx.x * blockDim.x + threadIdx.x;
    int stride = gridDim.x * blockDim.x;
    for (; i < n; i += stride) out[i] = (bf16_t)in[i];
}

// ---------------------------------------------------------------------------
// Gating: exact f32 path (sigmoid + bias, group-limited top-k, norm, *2.5).
// One wave per token; lanes 0..15 own expert e, lane pairs split H.
// Writes dense combine[N][E].
// ---------------------------------------------------------------------------
__global__ void k_gate(const float* __restrict__ x, const float* __restrict__ gk,
                       const float* __restrict__ gb, float* __restrict__ combine) {
    const int lane  = threadIdx.x & 31;
    const int wave  = threadIdx.x >> 5;
    const int token = blockIdx.x * (blockDim.x >> 5) + wave;
    if (token >= NTOK) return;

    const int e    = lane & 15;
    const int half = lane >> 4;

    const float* xr  = x  + (size_t)token * HDIM + half * 512;
    const float* gkp = gk + (size_t)half * 512 * NEXP + e;
    float acc = 0.f;
#pragma unroll 8
    for (int h = 0; h < 512; ++h) acc += xr[h] * gkp[h * NEXP];
    acc += __shfl_down(acc, 16, 32);           // lanes 0..15 hold full logits

    float sig = 1.f / (1.f + expf(-acc));      // sigmoid score
    float s4v = sig + gb[e];                   // score + correction bias

    float s4[16];
#pragma unroll
    for (int j = 0; j < 16; ++j) s4[j] = __shfl(s4v, j, 32);

    // group score = sum of top-2 within each group of 4
    float gs[4];
#pragma unroll
    for (int g = 0; g < 4; ++g) {
        float a = s4[4*g], b = s4[4*g+1], c = s4[4*g+2], d = s4[4*g+3];
        float hi1 = fmaxf(a, b), lo1 = fminf(a, b);
        float hi2 = fmaxf(c, d), lo2 = fminf(c, d);
        float m1 = fmaxf(hi1, hi2);
        float m2 = fmaxf(fminf(hi1, hi2), (hi1 >= hi2) ? lo1 : lo2);
        gs[g] = m1 + m2;
    }
    // top-2 groups, lower index wins ties (jax.lax.top_k semantics)
    bool gsel[4];
#pragma unroll
    for (int g = 0; g < 4; ++g) {
        int rank = 0;
#pragma unroll
        for (int g2 = 0; g2 < 4; ++g2)
            rank += (gs[g2] > gs[g]) || ((gs[g2] == gs[g]) && (g2 < g));
        gsel[g] = (rank < 2);
    }
    float mval[16];
#pragma unroll
    for (int j = 0; j < 16; ++j) mval[j] = gsel[j >> 2] ? s4[j] : 0.f;

    float denom = 1e-20f;
    bool  mysel = false;
#pragma unroll
    for (int j = 0; j < 16; ++j) {
        int rank = 0;
#pragma unroll
        for (int j2 = 0; j2 < 16; ++j2)
            rank += (mval[j2] > mval[j]) || ((mval[j2] == mval[j]) && (j2 < j));
        if (rank < 4) denom += mval[j];
        if (j == e)   mysel = (rank < 4);
    }
    float w = mysel ? (mval[e] / denom) * SCALEF : 0.f;
    if (lane < 16) combine[(size_t)token * NEXP + e] = w;
}

// ---------------------------------------------------------------------------
// WMMA tile loaders for V_WMMA_F32_16X16X32_BF16 (wave32).
// A (16x32, row-major [rows][ld]): lane p(+16*hi) = row M=p;
//   elems 0..7 = K k0+8*hi.. ; elems 8..15 = K k0+16+8*hi..
// B (32x16, row-major [K][ld], cols at n0): lane p(+16*hi) = col N=n0+p;
//   elems 0..7 = K k0+8*hi.. ; elems 8..15 = K k0+16+8*hi.. (strided by ld).
// ---------------------------------------------------------------------------
__device__ __forceinline__ v16bf load_a_tile(const bf16_t* __restrict__ base,
                                             int ld, int p, int hi, int k0) {
    const bf16_t* rp = base + (size_t)p * ld + k0 + hi * 8;
    v8bf lo = *(const v8bf*)(rp);
    v8bf hh = *(const v8bf*)(rp + 16);
    return __builtin_shufflevector(lo, hh, 0,1,2,3,4,5,6,7,8,9,10,11,12,13,14,15);
}

__device__ __forceinline__ v16bf load_b_tile(const bf16_t* __restrict__ base,
                                             int ld, int lane, int k0) {
    const bf16_t* cp = base + (size_t)(k0 + (lane >> 4) * 8) * ld + (lane & 15);
    v16bf r;
#pragma unroll
    for (int i = 0; i < 8; ++i) {
        r[i]     = cp[(size_t)i * ld];
        r[8 + i] = cp[(size_t)(16 + i) * ld];
    }
    return r;
}

__device__ __forceinline__ v8f wmma_bf16(v16bf a, v16bf b, v8f c) {
    return __builtin_amdgcn_wmma_f32_16x16x32_bf16(false, a, false, b,
                                                   (short)0, c, false, false);
}

// ---------------------------------------------------------------------------
// inter[e][n][m] = combine[n][e] * silu(x @ w_gate[e]) * (x @ w_up[e]) (e<16)
// inter_s[n][m2] =                silu(x @ sw_gate)    * (x @ sw_up)   (z==16)
// One wave computes a 32x32 tile (2 row-tiles x 2 col-tiles x {g,u}).
// ---------------------------------------------------------------------------
__global__ void k_inter(const bf16_t* __restrict__ xb,
                        const bf16_t* __restrict__ wg, const bf16_t* __restrict__ wu,
                        const bf16_t* __restrict__ swg, const bf16_t* __restrict__ swu,
                        const float* __restrict__ combine,
                        bf16_t* __restrict__ inter, bf16_t* __restrict__ inter_s) {
    const int lane = threadIdx.x & 31;
    const int wave = threadIdx.x >> 5;
    const int p    = lane & 15;
    const int hi   = lane >> 4;
    const int e    = blockIdx.z;
    const int c0   = (blockIdx.x * 4 + wave) * 32;   // 32 columns per wave
    const int n0   = blockIdx.y * 32;                // 32 rows per block

    const bf16_t *Bg, *Bu;
    bf16_t* out;
    int ldb, ldo;
    if (e < NEXP) {
        if (c0 >= MDIM) return;                      // wave-uniform exit
        Bg  = wg + (size_t)e * HDIM * MDIM + c0;
        Bu  = wu + (size_t)e * HDIM * MDIM + c0;
        out = inter + ((size_t)e * NTOK + n0) * MDIM + c0;
        ldb = MDIM; ldo = MDIM;
    } else {                                         // shared expert, 2M columns
        Bg  = swg + c0;
        Bu  = swu + c0;
        out = inter_s + (size_t)n0 * TWOM + c0;
        ldb = TWOM; ldo = TWOM;
    }
    const bf16_t* A = xb + (size_t)n0 * HDIM;

    v8f ag[2][2], au[2][2];
#pragma unroll
    for (int rt = 0; rt < 2; ++rt)
#pragma unroll
        for (int ct = 0; ct < 2; ++ct) {
            ag[rt][ct] = (v8f){0.f,0.f,0.f,0.f,0.f,0.f,0.f,0.f};
            au[rt][ct] = (v8f){0.f,0.f,0.f,0.f,0.f,0.f,0.f,0.f};
        }

    for (int k0 = 0; k0 < HDIM; k0 += 32) {
        v16bf a0 = load_a_tile(A, HDIM, p, hi, k0);
        v16bf a1 = load_a_tile(A + 16 * HDIM, HDIM, p, hi, k0);
#pragma unroll
        for (int ct = 0; ct < 2; ++ct) {
            v16bf bg = load_b_tile(Bg + ct * 16, ldb, lane, k0);
            v16bf bu = load_b_tile(Bu + ct * 16, ldb, lane, k0);
            ag[0][ct] = wmma_bf16(a0, bg, ag[0][ct]);
            ag[1][ct] = wmma_bf16(a1, bg, ag[1][ct]);
            au[0][ct] = wmma_bf16(a0, bu, au[0][ct]);
            au[1][ct] = wmma_bf16(a1, bu, au[1][ct]);
        }
    }

#pragma unroll
    for (int rt = 0; rt < 2; ++rt)
#pragma unroll
        for (int r = 0; r < 8; ++r) {
            const int row = n0 + rt * 16 + r + hi * 8;
            const float cw = (e < NEXP) ? combine[(size_t)row * NEXP + e] : 1.f;
#pragma unroll
            for (int ct = 0; ct < 2; ++ct) {
                float g = ag[rt][ct][r], u = au[rt][ct][r];
                float v = (g / (1.f + expf(-g))) * u * cw;  // silu(g)*u*combine
                out[((size_t)(rt * 16 + r + hi * 8)) * ldo + ct * 16 + p] = (bf16_t)v;
            }
        }
}

// ---------------------------------------------------------------------------
// out[n][h] = (inter_s @ sw_down)[n][h] + sum_e (scaled inter_e @ w_down[e])
// One long-K accumulation (K = 1024 + 16*512 = 9216) into a 32x64 wave tile.
// ---------------------------------------------------------------------------
__global__ void k_down(const bf16_t* __restrict__ inter, const bf16_t* __restrict__ inter_s,
                       const bf16_t* __restrict__ wd, const bf16_t* __restrict__ swd,
                       float* __restrict__ out) {
    const int lane = threadIdx.x & 31;
    const int wave = threadIdx.x >> 5;
    const int p    = lane & 15;
    const int hi   = lane >> 4;
    const int h0   = blockIdx.x * 64;                 // 64 output columns per wave
    const int n0   = blockIdx.y * 128 + wave * 32;    // 32 rows per wave

    v8f acc[2][4];
#pragma unroll
    for (int rt = 0; rt < 2; ++rt)
#pragma unroll
        for (int ct = 0; ct < 4; ++ct)
            acc[rt][ct] = (v8f){0.f,0.f,0.f,0.f,0.f,0.f,0.f,0.f};

    for (int seg = 0; seg < NEXP + 1; ++seg) {
        const bf16_t *A, *B;
        int lda, K;
        if (seg == 0) {                               // shared expert
            A = inter_s + (size_t)n0 * TWOM; lda = TWOM;
            B = swd + h0; K = TWOM;
        } else {
            const int e = seg - 1;
            A = inter + ((size_t)e * NTOK + n0) * MDIM; lda = MDIM;
            B = wd + (size_t)e * MDIM * HDIM + h0; K = MDIM;
        }
        for (int k0 = 0; k0 < K; k0 += 32) {
            v16bf a0 = load_a_tile(A, lda, p, hi, k0);
            v16bf a1 = load_a_tile(A + 16 * lda, lda, p, hi, k0);
#pragma unroll
            for (int ct = 0; ct < 4; ++ct) {
                v16bf b = load_b_tile(B + ct * 16, HDIM, lane, k0);
                acc[0][ct] = wmma_bf16(a0, b, acc[0][ct]);
                acc[1][ct] = wmma_bf16(a1, b, acc[1][ct]);
            }
        }
    }

#pragma unroll
    for (int rt = 0; rt < 2; ++rt)
#pragma unroll
        for (int ct = 0; ct < 4; ++ct)
#pragma unroll
            for (int r = 0; r < 8; ++r)
                out[(size_t)(n0 + rt * 16 + r + hi * 8) * HDIM + h0 + ct * 16 + p] =
                    acc[rt][ct][r];
}

// ---------------------------------------------------------------------------
extern "C" void kernel_launch(void* const* d_in, const int* in_sizes, int n_in,
                              void* d_out, int out_size, void* d_ws, size_t ws_size,
                              hipStream_t stream) {
    const float* x   = (const float*)d_in[0];   // [2048,1024]
    const float* gk  = (const float*)d_in[1];   // [1024,16]
    const float* gb  = (const float*)d_in[2];   // [16]
    const float* wg  = (const float*)d_in[3];   // [16,1024,512]
    const float* wu  = (const float*)d_in[4];   // [16,1024,512]
    const float* wd  = (const float*)d_in[5];   // [16,512,1024]
    const float* swg = (const float*)d_in[6];   // [1024,1024]
    const float* swu = (const float*)d_in[7];   // [1024,1024]
    const float* swd = (const float*)d_in[8];   // [1024,1024]
    float* out = (float*)d_out;                 // [2048,1024]

    char* ws = (char*)d_ws;
    bf16_t* xb   = (bf16_t*)(ws + OFF_X);
    bf16_t* wgb  = (bf16_t*)(ws + OFF_WG);
    bf16_t* wub  = (bf16_t*)(ws + OFF_WU);
    bf16_t* wdb  = (bf16_t*)(ws + OFF_WD);
    bf16_t* swgb = (bf16_t*)(ws + OFF_SWG);
    bf16_t* swub = (bf16_t*)(ws + OFF_SWU);
    bf16_t* swdb = (bf16_t*)(ws + OFF_SWD);
    bf16_t* inter   = (bf16_t*)(ws + OFF_INTER);
    bf16_t* inter_s = (bf16_t*)(ws + OFF_INTERS);
    float*  combine = (float*)(ws + OFF_COMB);

    // gating first: combine[] is consumed by k_inter (weights folded into A)
    k_gate<<<NTOK / 4, 128, 0, stream>>>(x, gk, gb, combine);

    const int CB = 256, CG = 2048;              // grid-stride converts
    k_convert<<<CG, CB, 0, stream>>>(x,   xb,   NTOK * HDIM);
    k_convert<<<CG, CB, 0, stream>>>(wg,  wgb,  NEXP * HDIM * MDIM);
    k_convert<<<CG, CB, 0, stream>>>(wu,  wub,  NEXP * HDIM * MDIM);
    k_convert<<<CG, CB, 0, stream>>>(wd,  wdb,  NEXP * MDIM * HDIM);
    k_convert<<<CG, CB, 0, stream>>>(swg, swgb, HDIM * TWOM);
    k_convert<<<CG, CB, 0, stream>>>(swu, swub, HDIM * TWOM);
    k_convert<<<CG, CB, 0, stream>>>(swd, swdb, TWOM * HDIM);

    // 32 cols/wave * 4 waves = 128 cols/block: 8 x-blocks cover shared 1024,
    // experts (512 cols) use the first 4 and early-out on the rest.
    dim3 gi(8, NTOK / 32, NEXP + 1);
    k_inter<<<gi, 128, 0, stream>>>(xb, wgb, wub, swgb, swub, combine, inter, inter_s);

    dim3 gd(HDIM / 64, NTOK / 128);
    k_down<<<gd, 128, 0, stream>>>(inter, inter_s, wdb, swdb, out);

    (void)in_sizes; (void)n_in; (void)out_size; (void)ws_size;
}
